// SWAttention_29944511988524
// MI455X (gfx1250) — matrix operand order repaired and checked by reading, not compiled, not run on hardware
//
#include <hip/hip_runtime.h>
#include <math.h>

#define D_    1024
#define T_    2048
#define B_    2
#define H_    16
#define HD_   64
#define WIN_  256
#define MTOT  (B_*T_)      // 4096
#define NCHUNK 17          // covers [q0-272, q0+271] span of 544 keys

typedef __attribute__((ext_vector_type(16))) __bf16 v16bf;
typedef __attribute__((ext_vector_type(8)))  float  v8f;
typedef unsigned short bfraw;

union Frag16 {
    v16bf v;
    uint4 q[2];
    bfraw u[16];
};

__device__ __forceinline__ bfraw f2bf(float f) {
    unsigned x = __float_as_uint(f);
    unsigned r = x + 0x7FFFu + ((x >> 16) & 1u);
    return (bfraw)(r >> 16);
}

__device__ __forceinline__ v8f wmma_bf16(v16bf a, v16bf b, v8f c) {
    return __builtin_amdgcn_wmma_f32_16x16x32_bf16(false, a, false, b, (short)0, c, false, false);
}

// ---------------- weight fp32 (K,N) -> bf16 transposed (N,K) ----------------
__global__ void cvtT_kernel(const float* __restrict__ src, bfraw* __restrict__ dst) {
    int i = blockIdx.x * blockDim.x + threadIdx.x;   // over D*D, i = n*1024 + k
    int n = i >> 10;
    int k = i & (D_ - 1);
    dst[i] = f2bf(src[(size_t)k * D_ + n]);
}

// ---------------- x (D,T,B) -> xbf (B*T, D) bf16 ----------------
__global__ void xpose_kernel(const float* __restrict__ x, bfraw* __restrict__ xbf) {
    int i  = blockIdx.x * blockDim.x + threadIdx.x;   // over MTOT*D
    int d  = i & (D_ - 1);
    int bt = i >> 10;
    int b  = bt >> 11;           // / T_
    int t  = bt & (T_ - 1);
    xbf[i] = f2bf(x[(size_t)d * (T_ * B_) + t * B_ + b]);
}

// ---------------- GEMM: C[M=4096, N=1024] = A[M,1024] * Wt^T + bias ----------------
// Wt is bf16 (N=1024, K=1024) row-major (pre-transposed weight).
// mode 0: bf16 row-major (M x 1024)            -> Q, K
// mode 1: fp32 transposed (D,T,B)              -> final output projection
// mode 2: bf16 transposed (b*1024+col)*T + t   -> V as (B,H,hd,T)
__global__ __launch_bounds__(256) void gemm_kernel(
    const bfraw* __restrict__ A, const bfraw* __restrict__ Wt,
    const float* __restrict__ bias,
    bfraw* __restrict__ out_bf, float* __restrict__ out_f, int mode)
{
    __shared__ __align__(16) bfraw As[64 * 40];    // 64 rows (M) x 32 K, padded stride 40
    __shared__ __align__(16) bfraw Bs[128 * 40];   // 128 cols (N) x 32 K, padded stride 40

    const int tid   = threadIdx.x;
    const int lane  = tid & 31;
    const int wave  = tid >> 5;
    const int waveM = wave >> 2;    // 0..1 : 32 rows each
    const int waveN = wave & 3;     // 0..3 : 32 cols each
    const int m0 = blockIdx.y * 64;
    const int n0 = blockIdx.x * 128;

    const int la    = lane & 15;
    const int koffA = (lane & 16);        // A frag: K element offset 0 / 16
    const int koffB = (lane & 16);        // B frag: K element offset 0 / 16

    const int arow = tid >> 2;            // 0..63
    const int acol = (tid & 3) * 8;       // 0,8,16,24
    const int brow = tid >> 1;            // 0..127 (n)
    const int bcol = (tid & 1) * 16;      // 0 / 16 (k)

    v8f acc[2][2];
    #pragma unroll
    for (int mi = 0; mi < 2; ++mi)
        #pragma unroll
        for (int ni = 0; ni < 2; ++ni)
            acc[mi][ni] = (v8f)0.0f;

    for (int k0 = 0; k0 < 1024; k0 += 32) {
        // stage A tile (64 x 32), contiguous b128 both ways
        uint4 av = *(const uint4*)(A + (size_t)(m0 + arow) * 1024 + k0 + acol);
        *(uint4*)(As + arow * 40 + acol) = av;
        // stage B tile (128n x 32k) from pre-transposed weights, contiguous b128
        uint4 bv0 = *(const uint4*)(Wt + (size_t)(n0 + brow) * 1024 + k0 + bcol);
        uint4 bv1 = *(const uint4*)(Wt + (size_t)(n0 + brow) * 1024 + k0 + bcol + 8);
        *(uint4*)(Bs + brow * 40 + bcol)     = bv0;
        *(uint4*)(Bs + brow * 40 + bcol + 8) = bv1;
        if (k0 + 32 < 1024) {
            __builtin_prefetch(A + (size_t)(m0 + arow) * 1024 + k0 + 32 + acol, 0, 0);
            __builtin_prefetch(Wt + (size_t)(n0 + brow) * 1024 + k0 + 32 + bcol, 0, 0);
        }
        __syncthreads();

        Frag16 af[2], bf[2];
        #pragma unroll
        for (int mi = 0; mi < 2; ++mi) {
            int r = waveM * 32 + mi * 16 + la;
            af[mi].q[0] = *(const uint4*)(As + r * 40 + koffA);
            af[mi].q[1] = *(const uint4*)(As + r * 40 + koffA + 8);
        }
        #pragma unroll
        for (int ni = 0; ni < 2; ++ni) {
            int c = waveN * 32 + ni * 16 + la;
            bf[ni].q[0] = *(const uint4*)(Bs + c * 40 + koffB);
            bf[ni].q[1] = *(const uint4*)(Bs + c * 40 + koffB + 8);
        }
        #pragma unroll
        for (int mi = 0; mi < 2; ++mi)
            #pragma unroll
            for (int ni = 0; ni < 2; ++ni)
                acc[mi][ni] = wmma_bf16(af[mi].v, bf[ni].v, acc[mi][ni]);
        __syncthreads();
    }

    const int rbase = (lane >> 4) * 8;
    #pragma unroll
    for (int mi = 0; mi < 2; ++mi) {
        #pragma unroll
        for (int ni = 0; ni < 2; ++ni) {
            int col = n0 + waveN * 32 + ni * 16 + la;
            float bv = bias[col];
            #pragma unroll
            for (int r = 0; r < 8; ++r) {
                int row = m0 + waveM * 32 + mi * 16 + rbase + r;
                float v = acc[mi][ni][r] + bv;
                if (mode == 0) {
                    out_bf[(size_t)row * 1024 + col] = f2bf(v);
                } else if (mode == 1) {
                    int b = row >> 11;
                    int t = row & (T_ - 1);
                    out_f[(size_t)col * (T_ * B_) + t * B_ + b] = v;
                } else {
                    int b = row >> 11;
                    int t = row & (T_ - 1);
                    out_bf[((size_t)(b * 1024 + col)) * T_ + t] = f2bf(v);
                }
            }
        }
    }
}

// ---------------- sliding-window sig-softmax attention ----------------
// Q/K: (B*T, D) bf16 row-major.  Vt: (B,H,hd,T) bf16.  One wave per 16-query tile.
__global__ __launch_bounds__(128) void attn_kernel(
    const bfraw* __restrict__ Qb, const bfraw* __restrict__ Kb,
    const bfraw* __restrict__ Vt, bfraw* __restrict__ Ob)
{
    __shared__ __align__(16) bfraw ps[4][16 * 40];   // per-wave P tile (16q x 32k), stride 40

    const int tid  = threadIdx.x;
    const int lane = tid & 31;
    const int w    = tid >> 5;
    const int qtile = blockIdx.x * 4 + w;       // 0 .. B*H*(T/16)-1
    const int bh = qtile >> 7;                  // / (T/16 = 128)
    const int qt = qtile & 127;
    const int b  = bh >> 4;
    const int h  = bh & 15;
    const int q0 = qt * 16;

    const int la    = lane & 15;
    const int koffA = (lane & 16);
    const int koffB = (lane & 16);
    const int rbase = (lane >> 4) * 8;
    const size_t headoff = (size_t)h * HD_;

    // load Q fragments (16 rows x 64 dims -> 2 k-steps of 32)
    Frag16 aq[2];
    {
        size_t base = ((size_t)(b * T_ + q0 + la)) * D_ + headoff;
        #pragma unroll
        for (int ds = 0; ds < 2; ++ds) {
            aq[ds].q[0] = *(const uint4*)(Qb + base + ds * 32 + koffA);
            aq[ds].q[1] = *(const uint4*)(Qb + base + ds * 32 + koffA + 8);
        }
    }

    v8f vo[4];
    #pragma unroll
    for (int nt = 0; nt < 4; ++nt) vo[nt] = (v8f)0.0f;
    float mrun[8], lrun[8];
    #pragma unroll
    for (int r = 0; r < 8; ++r) { mrun[r] = -3e38f; lrun[r] = 0.0f; }

    int kstart = q0 - WIN_;
    if (kstart < 0) kstart = 0;
    kstart &= ~31;

    for (int j = 0; j < NCHUNK; ++j) {
        const int kc = kstart + j * 32;

        // ---- scores S = Q * K^T for 32 keys (two 16-col C tiles) ----
        v8f sc[2];
        sc[0] = (v8f)0.0f; sc[1] = (v8f)0.0f;
        #pragma unroll
        for (int half = 0; half < 2; ++half) {
            int key  = kc + half * 16 + la;
            int keyc = key < (T_ - 1) ? key : (T_ - 1);
            size_t krow = ((size_t)(b * T_ + keyc)) * D_ + headoff;
            #pragma unroll
            for (int ds = 0; ds < 2; ++ds) {
                Frag16 bk;
                bk.q[0] = *(const uint4*)(Kb + krow + ds * 32 + koffB);
                bk.q[1] = *(const uint4*)(Kb + krow + ds * 32 + koffB + 8);
                sc[half] = wmma_bf16(aq[ds].v, bk.v, sc[half]);
            }
        }

        // ---- masked sig-softmax (online, per-row stats in C layout) ----
        float p0[8], p1[8], scl[8];
        #pragma unroll
        for (int r = 0; r < 8; ++r) {
            const int qrow = q0 + rbase + r;
            float y0, y1;
            {
                int key = kc + la;
                float s = sc[0][r] * 0.125f;
                int dd = qrow - key; if (dd < 0) dd = -dd;
                y0 = (key >= T_ || dd > WIN_) ? -3e38f
                     : (s - __logf(1.0f + __expf(-s)));
            }
            {
                int key = kc + 16 + la;
                float s = sc[1][r] * 0.125f;
                int dd = qrow - key; if (dd < 0) dd = -dd;
                y1 = (key >= T_ || dd > WIN_) ? -3e38f
                     : (s - __logf(1.0f + __expf(-s)));
            }
            float cm = fmaxf(y0, y1);
            cm = fmaxf(cm, __shfl_xor(cm, 1, 32));
            cm = fmaxf(cm, __shfl_xor(cm, 2, 32));
            cm = fmaxf(cm, __shfl_xor(cm, 4, 32));
            cm = fmaxf(cm, __shfl_xor(cm, 8, 32));
            float mn  = fmaxf(mrun[r], cm);
            float rsc = __expf(mrun[r] - mn);
            float e0 = (y0 < -1e37f) ? 0.0f : __expf(y0 - mn);
            float e1 = (y1 < -1e37f) ? 0.0f : __expf(y1 - mn);
            float ls = e0 + e1;
            ls += __shfl_xor(ls, 1, 32);
            ls += __shfl_xor(ls, 2, 32);
            ls += __shfl_xor(ls, 4, 32);
            ls += __shfl_xor(ls, 8, 32);
            lrun[r] = lrun[r] * rsc + ls;
            mrun[r] = mn;
            p0[r] = e0; p1[r] = e1; scl[r] = rsc;
        }
        #pragma unroll
        for (int nt = 0; nt < 4; ++nt)
            #pragma unroll
            for (int r = 0; r < 8; ++r)
                vo[nt][r] *= scl[r];

        // ---- P tile to LDS (C layout -> row-major bf16), then A fragment ----
        #pragma unroll
        for (int r = 0; r < 8; ++r) {
            ps[w][(rbase + r) * 40 + la]      = f2bf(p0[r]);
            ps[w][(rbase + r) * 40 + 16 + la] = f2bf(p1[r]);
        }
        Frag16 ap;
        ap.q[0] = *(const uint4*)(&ps[w][la * 40 + koffA]);
        ap.q[1] = *(const uint4*)(&ps[w][la * 40 + koffA + 8]);

        // ---- O += P * V : V is (B,H,hd,T), keys contiguous -> b128 loads ----
        #pragma unroll
        for (int nt = 0; nt < 4; ++nt) {
            size_t vrow = ((size_t)(b * 1024 + h * 64 + nt * 16 + la)) * T_;
            Frag16 bvv;
            bvv.q[0] = *(const uint4*)(Vt + vrow + kc + koffB);
            bvv.q[1] = *(const uint4*)(Vt + vrow + kc + koffB + 8);
            vo[nt] = wmma_bf16(ap.v, bvv.v, vo[nt]);
        }
    }

    // ---- finalize: divide by running sum, store bf16 ----
    #pragma unroll
    for (int nt = 0; nt < 4; ++nt) {
        #pragma unroll
        for (int r = 0; r < 8; ++r) {
            float val = vo[nt][r] / lrun[r];
            Ob[((size_t)(b * T_ + q0 + rbase + r)) * D_ + headoff + nt * 16 + la] = f2bf(val);
        }
    }
}

// ---------------- host launcher ----------------
extern "C" void kernel_launch(void* const* d_in, const int* in_sizes, int n_in,
                              void* d_out, int out_size, void* d_ws, size_t ws_size,
                              hipStream_t stream) {
    const float* x  = (const float*)d_in[0];
    const float* Wq = (const float*)d_in[1];
    const float* bq = (const float*)d_in[2];
    const float* Wk = (const float*)d_in[3];
    const float* bk = (const float*)d_in[4];
    const float* Wv = (const float*)d_in[5];
    const float* bv = (const float*)d_in[6];
    const float* Wo = (const float*)d_in[7];
    const float* bo = (const float*)d_in[8];
    float* out = (float*)d_out;

    char* ws = (char*)d_ws;
    // byte offsets (total 48 MiB)
    bfraw* xbf = (bfraw*)(ws + 0);                 //  8 MiB  (B*T, D) bf16
    bfraw* wqb = (bfraw*)(ws + (8u  << 20));       //  2 MiB  transposed (N,K)
    bfraw* wkb = (bfraw*)(ws + (10u << 20));       //  2 MiB
    bfraw* wvb = (bfraw*)(ws + (12u << 20));       //  2 MiB
    bfraw* wob = (bfraw*)(ws + (14u << 20));       //  2 MiB
    bfraw* qb  = (bfraw*)(ws + (16u << 20));       //  8 MiB  (B*T, D)
    bfraw* kb  = (bfraw*)(ws + (24u << 20));       //  8 MiB  (B*T, D)
    bfraw* vbT = (bfraw*)(ws + (32u << 20));       //  8 MiB  (B,H,hd,T)  (not last: OOB-slack)
    bfraw* ab  = (bfraw*)(ws + (40u << 20));       //  8 MiB  (B*T, D)

    // convert inputs to bf16 (weights transposed to (N,K))
    xpose_kernel<<<(MTOT * D_) / 256, 256, 0, stream>>>(x, xbf);
    cvtT_kernel<<<(D_ * D_) / 256, 256, 0, stream>>>(Wq, wqb);
    cvtT_kernel<<<(D_ * D_) / 256, 256, 0, stream>>>(Wk, wkb);
    cvtT_kernel<<<(D_ * D_) / 256, 256, 0, stream>>>(Wv, wvb);
    cvtT_kernel<<<(D_ * D_) / 256, 256, 0, stream>>>(Wo, wob);

    // QKV projections (V written transposed for attention B-fragments)
    dim3 ggrid(1024 / 128, MTOT / 64);
    gemm_kernel<<<ggrid, 256, 0, stream>>>(xbf, wqb, bq, qb, nullptr, 0);
    gemm_kernel<<<ggrid, 256, 0, stream>>>(xbf, wkb, bk, kb, nullptr, 0);
    gemm_kernel<<<ggrid, 256, 0, stream>>>(xbf, wvb, bv, vbT, nullptr, 2);

    // attention: one wave per 16-query tile, 4 waves per block
    attn_kernel<<<(B_ * H_ * (T_ / 16)) / 4, 128, 0, stream>>>(qb, kb, vbT, ab);

    // output projection, write fp32 (D,T,B)
    gemm_kernel<<<ggrid, 256, 0, stream>>>(ab, wob, bo, nullptr, out, 1);
}
